// NonAutoRegressive_67233418052402
// MI455X (gfx1250) — matrix operand order repaired and checked by reading, not compiled, not run on hardware
//
#include <hip/hip_runtime.h>
#include <hip/hip_bf16.h>

#define NN 100000
#define EE 400000
#define DD 128
#define LL 4
#define BN_EPS 1e-5f
#define AGG_EPS 1e-6f
#define WPITCH 136   // LDS row pitch (bf16 elems): 272B = 68 banks -> conflict-free

typedef __attribute__((ext_vector_type(16))) __bf16 v16bf;
typedef __attribute__((ext_vector_type(8)))  __bf16 v8bf;
typedef __attribute__((ext_vector_type(8)))  float  v8f;
typedef __attribute__((ext_vector_type(4)))  float  v4f;

// ---------------------------------------------------------------------------
// Generic zero
// ---------------------------------------------------------------------------
__global__ void zero_f32(float* __restrict__ p, long n) {
    long i = (long)blockIdx.x * blockDim.x + threadIdx.x;
    long stride = (long)gridDim.x * blockDim.x;
    for (; i < n; i += stride) p[i] = 0.0f;
}

// ---------------------------------------------------------------------------
// Reduce overlap_length -> sum, sumsq  (red[0], red[1])
// ---------------------------------------------------------------------------
__global__ void reduce_len(const float* __restrict__ len, float* __restrict__ red, int E) {
    __shared__ float s0[256], s1[256];
    int t = threadIdx.x;
    float a = 0.0f, b = 0.0f;
    for (int i = blockIdx.x * 256 + t; i < E; i += gridDim.x * 256) {
        float x = len[i];
        a += x; b += x * x;
    }
    s0[t] = a; s1[t] = b;
    __syncthreads();
    for (int o = 128; o > 0; o >>= 1) {
        if (t < o) { s0[t] += s0[t + o]; s1[t] += s1[t + o]; }
        __syncthreads();
    }
    if (t == 0) { atomicAdd(&red[0], s0[0]); atomicAdd(&red[1], s1[0]); }
}

// ---------------------------------------------------------------------------
// Edge encoder (float4 per thread): e = [sim, (len-mu)/std_unbiased]@W_enc+b
// ---------------------------------------------------------------------------
__global__ void encode_edges(const float* __restrict__ sim, const float* __restrict__ len,
                             const float* __restrict__ red,
                             const float* __restrict__ W_enc, const float* __restrict__ b_enc,
                             float* __restrict__ e, int E) {
    long t = (long)blockIdx.x * blockDim.x + threadIdx.x;
    long total4 = (long)E * (DD / 4);
    if (t >= total4) return;
    long idx4 = t * 4;
    int i = (int)(idx4 >> 7);
    int c = (int)(idx4 & (DD - 1));
    float Ef = (float)E;
    float mean = red[0] / Ef;
    float var  = (red[1] - Ef * mean * mean) / (Ef - 1.0f);  // unbiased (torch.std)
    float el = (len[i] - mean) * rsqrtf(var);
    float s  = sim[i];
    v4f w0 = *(const v4f*)(W_enc + c);
    v4f w1 = *(const v4f*)(W_enc + DD + c);
    v4f bb = *(const v4f*)(b_enc + c);
    v4f r;
    r.x = s * w0.x + el * w1.x + bb.x;
    r.y = s * w0.y + el * w1.y + bb.y;
    r.z = s * w0.z + el * w1.z + bb.z;
    r.w = s * w0.w + el * w1.w + bb.w;
    *(v4f*)(e + idx4) = r;
}

// ---------------------------------------------------------------------------
// h = ones
// ---------------------------------------------------------------------------
__global__ void init_h(float* __restrict__ h, long n) {
    long i = (long)blockIdx.x * blockDim.x + threadIdx.x;
    long stride = (long)gridDim.x * blockDim.x;
    for (; i < n; i += stride) h[i] = 1.0f;
}

// ---------------------------------------------------------------------------
// WMMA GEMM: Y[M,128] = X[M,128] @ W[128,128] + b
// Block = 512 threads = 16 waves; W staged once per block into LDS as bf16,
// TRANSPOSED with padded pitch (Wt[n*136+k]) so each lane's B-fragment
// ds_load_b128 pair tiles all 64 banks (start bank = 4*lm + const).
// Each wave computes a 16x128 strip: 8 N-tiles x 4 K-chunks of
// v_wmma_f32_16x16x32_bf16 (f32 accumulate).
// ---------------------------------------------------------------------------
__global__ void __launch_bounds__(512, 1)
wmma_lin(const float* __restrict__ X, const float* __restrict__ W,
         const float* __restrict__ b, float* __restrict__ Y, int M) {
    __shared__ __bf16 Wt[DD * WPITCH];         // ~34 KB

    // cooperative stage: fp32 row-major W[k][n] -> bf16 transposed Wt[n][k]
    for (int i = threadIdx.x; i < DD * DD; i += blockDim.x) {
        int k = i >> 7, n = i & (DD - 1);
        Wt[n * WPITCH + k] = (__bf16)W[i];
    }
    __syncthreads();

    const int wave = blockIdx.x * (blockDim.x >> 5) + (threadIdx.x >> 5);
    const int lane = threadIdx.x & 31;
    const int m0 = wave * 16;
    if (m0 >= M) return;                       // wave-uniform after the barrier
    const int half = lane >> 4;                // 0: lanes 0-15, 1: lanes 16-31
    const int lm   = lane & 15;

    v8f acc[8] = {};
    const float* xrow = X + (long)(m0 + lm) * DD;

#pragma unroll
    for (int kc = 0; kc < 4; ++kc) {
        const int kbase = kc * 32;
        // A fragment: two contiguous 8-float runs -> 4x global b128 loads
        const int abase = kbase + half * 8;
        v4f a0 = *(const v4f*)(xrow + abase);
        v4f a1 = *(const v4f*)(xrow + abase + 4);
        v4f a2 = *(const v4f*)(xrow + abase + 16);
        v4f a3 = *(const v4f*)(xrow + abase + 20);
        v16bf a = {};
#pragma unroll
        for (int j = 0; j < 4; ++j) {
            a[j]      = (__bf16)a0[j];
            a[j + 4]  = (__bf16)a1[j];
            a[j + 8]  = (__bf16)a2[j];
            a[j + 12] = (__bf16)a3[j];
        }
#pragma unroll
        for (int nt = 0; nt < 8; ++nt) {
            const int n0 = nt * 16;
            // B fragment: two 16B LDS reads (conflict-free with pitch 136)
            const v8bf* wp = (const v8bf*)(Wt + (n0 + lm) * WPITCH + kbase + half * 8);
            v8bf blo = wp[0];
            v8bf bhi = wp[2];
            v16bf bf = {};
#pragma unroll
            for (int j = 0; j < 8; ++j) {
                bf[j]     = blo[j];
                bf[j + 8] = bhi[j];
            }
            acc[nt] = __builtin_amdgcn_wmma_f32_16x16x32_bf16(
                false, a, false, bf, (short)0, acc[nt], false, false);
        }
    }

    // C/D f32 layout: VGPR r -> row r + 8*half, col lm
#pragma unroll
    for (int nt = 0; nt < 8; ++nt) {
        const int n0 = nt * 16;
        const float bias = b[n0 + lm];
#pragma unroll
        for (int r = 0; r < 8; ++r) {
            int row = m0 + r + half * 8;
            Y[(long)row * DD + n0 + lm] = acc[nt][r] + bias;
        }
    }
}

// ---------------------------------------------------------------------------
// Edge message pass (fused): ehat = Ce + Dh[src] + Eh[dst]; sigma = sigmoid;
// scatter num/den; accumulate BN batch stats for e_hat.
// Ce lives in the ehat buffer (read-then-overwrite per element, same thread).
// Thread = channel; 64-edge index window staged in LDS (broadcast reads).
// ---------------------------------------------------------------------------
#define EPB 64
__global__ void edge_message(float* __restrict__ ehat /* in: Ce, out: e_hat */,
                             const float* __restrict__ Dh, const float* __restrict__ Eh,
                             const float* __restrict__ Bh,
                             const int* __restrict__ src, const int* __restrict__ dst,
                             float* __restrict__ num, float* __restrict__ den,
                             float* __restrict__ stats, int E) {
    __shared__ int ssrc[EPB], sdst[EPB];
    const int c = threadIdx.x;               // 0..127
    const int e0 = blockIdx.x * EPB;
    const int cnt = ((e0 + EPB < E) ? EPB : (E - e0));
    if (c < EPB && c < cnt) { ssrc[c] = src[e0 + c]; sdst[c] = dst[e0 + c]; }
    __syncthreads();
    float s = 0.0f, ss = 0.0f;
    for (int i = 0; i < cnt; ++i) {
        int u = ssrc[i], v = sdst[i];
        long ei = (long)(e0 + i) * DD + c;
        float x = ehat[ei] + Dh[(long)u * DD + c] + Eh[(long)v * DD + c];
        float sg = 1.0f / (1.0f + expf(-x));
        ehat[ei] = x;
        atomicAdd(&num[(long)v * DD + c], sg * Bh[(long)u * DD + c]);
        atomicAdd(&den[(long)v * DD + c], sg);
        s += x; ss += x * x;
    }
    atomicAdd(&stats[c], s);          // sum_e
    atomicAdd(&stats[DD + c], ss);    // sumsq_e
}

// ---------------------------------------------------------------------------
// Node update: hnew = Ah + num/(den+eps) (written over num); BN stats for h.
// ---------------------------------------------------------------------------
#define NPB 64
__global__ void node_update(const float* __restrict__ Ah, float* __restrict__ num,
                            const float* __restrict__ den, float* __restrict__ stats, int N) {
    const int c = threadIdx.x;
    const int n0 = blockIdx.x * NPB;
    const int n1 = (n0 + NPB < N) ? n0 + NPB : N;
    float s = 0.0f, ss = 0.0f;
    for (int n = n0; n < n1; ++n) {
        long idx = (long)n * DD + c;
        float hn = Ah[idx] + num[idx] / (den[idx] + AGG_EPS);
        num[idx] = hn;
        s += hn; ss += hn * hn;
    }
    atomicAdd(&stats[256 + c], s);    // sum_h
    atomicAdd(&stats[384 + c], ss);   // sumsq_h
}

// ---------------------------------------------------------------------------
// Finalize BN stats: mean / gamma*rsqrt(var+eps) for e (t<128) and h (t>=128)
// stats: [0]sum_e [128]sumsq_e [256]sum_h [384]sumsq_h
//        [512]mean_e [640]scale_e [768]mean_h [896]scale_h
// ---------------------------------------------------------------------------
__global__ void finalize_stats(float* __restrict__ stats,
                               const float* __restrict__ gamma_e,
                               const float* __restrict__ gamma_h) {
    int t = threadIdx.x;
    if (t < 128) {
        float cnt = (float)EE;
        float mean = stats[t] / cnt;
        float var  = stats[128 + t] / cnt - mean * mean;   // biased (BN training)
        stats[512 + t] = mean;
        stats[640 + t] = gamma_e[t] * rsqrtf(var + BN_EPS);
    } else {
        int c = t - 128;
        float cnt = (float)NN;
        float mean = stats[256 + c] / cnt;
        float var  = stats[384 + c] / cnt - mean * mean;
        stats[768 + c] = mean;
        stats[896 + c] = gamma_h[c] * rsqrtf(var + BN_EPS);
    }
}

// ---------------------------------------------------------------------------
// X += relu((Xn - mean)*scale + beta)   (float4 per thread)
// ---------------------------------------------------------------------------
__global__ void apply_bnrelu(float* __restrict__ X, const float* __restrict__ Xn,
                             const float* __restrict__ mean, const float* __restrict__ scale,
                             const float* __restrict__ beta, long total4) {
    long t = (long)blockIdx.x * blockDim.x + threadIdx.x;
    if (t >= total4) return;
    long idx = t * 4;
    int c = (int)(idx & (DD - 1));
    v4f xn = *(const v4f*)(Xn + idx);
    v4f mu = *(const v4f*)(mean + c);
    v4f sc = *(const v4f*)(scale + c);
    v4f be = *(const v4f*)(beta + c);
    v4f x  = *(const v4f*)(X + idx);
    x.x += fmaxf((xn.x - mu.x) * sc.x + be.x, 0.0f);
    x.y += fmaxf((xn.y - mu.y) * sc.y + be.y, 0.0f);
    x.z += fmaxf((xn.z - mu.z) * sc.z + be.z, 0.0f);
    x.w += fmaxf((xn.w - mu.w) * sc.w + be.w, 0.0f);
    *(v4f*)(X + idx) = x;
}

// ---------------------------------------------------------------------------
// Decoder: out[i] = [h[src], h[dst], e[i]] . W_dec + b_dec
// 1 wave / edge; each lane owns 4 contiguous channels (b128 loads).
// ---------------------------------------------------------------------------
__global__ void decode(const float* __restrict__ h, const float* __restrict__ e,
                       const int* __restrict__ src, const int* __restrict__ dst,
                       const float* __restrict__ Wd, const float* __restrict__ bd,
                       float* __restrict__ out, int E) {
    const int wave = (blockIdx.x * blockDim.x + threadIdx.x) >> 5;
    const int lane = threadIdx.x & 31;
    if (wave >= E) return;
    int u = src[wave], v = dst[wave];
    const int c = lane * 4;
    v4f hu = *(const v4f*)(h + (long)u * DD + c);
    v4f hv = *(const v4f*)(h + (long)v * DD + c);
    v4f er = *(const v4f*)(e + (long)wave * DD + c);
    v4f w0 = *(const v4f*)(Wd + c);
    v4f w1 = *(const v4f*)(Wd + DD + c);
    v4f w2 = *(const v4f*)(Wd + 2 * DD + c);
    float acc = hu.x * w0.x + hu.y * w0.y + hu.z * w0.z + hu.w * w0.w
              + hv.x * w1.x + hv.y * w1.y + hv.z * w1.z + hv.w * w1.w
              + er.x * w2.x + er.y * w2.y + er.z * w2.z + er.w * w2.w;
    for (int off = 16; off > 0; off >>= 1)
        acc += __shfl_down(acc, off, 32);
    if (lane == 0) out[wave] = acc + bd[0];
}

// ---------------------------------------------------------------------------
// Host orchestration
// ---------------------------------------------------------------------------
extern "C" void kernel_launch(void* const* d_in, const int* in_sizes, int n_in,
                              void* d_out, int out_size, void* d_ws, size_t ws_size,
                              hipStream_t stream) {
    const int*   src   = (const int*)d_in[0];
    const int*   dst   = (const int*)d_in[1];
    const float* sim   = (const float*)d_in[2];
    const float* lenv  = (const float*)d_in[3];
    const float* W_enc = (const float*)d_in[5];
    const float* b_enc = (const float*)d_in[6];
    const float* W_A = (const float*)d_in[7];   const float* b_A = (const float*)d_in[8];
    const float* W_B = (const float*)d_in[9];   const float* b_B = (const float*)d_in[10];
    const float* W_C = (const float*)d_in[11];  const float* b_C = (const float*)d_in[12];
    const float* W_D = (const float*)d_in[13];  const float* b_D = (const float*)d_in[14];
    const float* W_E = (const float*)d_in[15];  const float* b_E = (const float*)d_in[16];
    const float* gamma_h = (const float*)d_in[17];
    const float* beta_h  = (const float*)d_in[18];
    const float* gamma_e = (const float*)d_in[19];
    const float* beta_e  = (const float*)d_in[20];
    const float* W_dec = (const float*)d_in[21];
    const float* b_dec = (const float*)d_in[22];
    float* out = (float*)d_out;

    const long N128 = (long)NN * DD;   // 12.8M floats
    const long E128 = (long)EE * DD;   // 51.2M floats

    float* ws    = (float*)d_ws;
    float* e     = ws;                 // [E,128]
    float* ehat  = e    + E128;        // [E,128]  Ce then e_hat
    float* h     = ehat + E128;        // [N,128]
    float* Ah    = h    + N128;
    float* Bh    = Ah   + N128;
    float* Dh    = Bh   + N128;
    float* Eh    = Dh   + N128;
    float* num   = Eh   + N128;        // also hnew after node_update
    float* den   = num  + N128;
    float* stats = den  + N128;        // 1024 floats (must follow den: zeroed together)
    float* red   = stats + 1024;       // 2 floats

    // --- preamble ---
    zero_f32<<<1, 256, 0, stream>>>(stats + 512, 514);              // means/scales + red
    reduce_len<<<256, 256, 0, stream>>>(lenv, red, EE);
    {
        long total4 = E128 / 4;
        int grid = (int)((total4 + 255) / 256);
        encode_edges<<<grid, 256, 0, stream>>>(sim, lenv, red, W_enc, b_enc, e, EE);
    }
    init_h<<<1024, 256, 0, stream>>>(h, N128);

    // 512 threads = 16 waves = 256 rows per block
    const int gemmN = (NN + 255) / 256;
    const int gemmE = (EE + 255) / 256;

    for (int l = 0; l < LL; ++l) {
        const long wo = (long)l * DD * DD;
        const long bo = (long)l * DD;

        // zero num, den, stats[0:512) (contiguous)
        zero_f32<<<2048, 256, 0, stream>>>(num, 2 * N128 + 512);

        // dense GEMMs via bf16 WMMA (LDS-staged transposed weights)
        wmma_lin<<<gemmN, 512, 0, stream>>>(h, W_A + wo, b_A + bo, Ah, NN);
        wmma_lin<<<gemmN, 512, 0, stream>>>(h, W_B + wo, b_B + bo, Bh, NN);
        wmma_lin<<<gemmN, 512, 0, stream>>>(h, W_D + wo, b_D + bo, Dh, NN);
        wmma_lin<<<gemmN, 512, 0, stream>>>(h, W_E + wo, b_E + bo, Eh, NN);
        wmma_lin<<<gemmE, 512, 0, stream>>>(e, W_C + wo, b_C + bo, ehat, EE); // Ce

        // fused gather + sigmoid + scatter + BN stats
        edge_message<<<EE / EPB, 128, 0, stream>>>(ehat, Dh, Eh, Bh, src, dst,
                                                   num, den, stats, EE);
        node_update<<<(NN + NPB - 1) / NPB, 128, 0, stream>>>(Ah, num, den, stats, NN);
        finalize_stats<<<1, 256, 0, stream>>>(stats, gamma_e + bo, gamma_h + bo);

        apply_bnrelu<<<(int)((N128 / 4 + 255) / 256), 256, 0, stream>>>(
            h, num, stats + 768, stats + 896, beta_h + bo, N128 / 4);
        apply_bnrelu<<<(int)((E128 / 4 + 255) / 256), 256, 0, stream>>>(
            e, ehat, stats + 512, stats + 640, beta_e + bo, E128 / 4);
    }

    // decoder: one wave per edge
    decode<<<(EE * 32 + 255) / 256, 256, 0, stream>>>(h, e, src, dst, W_dec, b_dec, out, EE);
}
